// ViT_6167573037156
// MI455X (gfx1250) — matrix operand order repaired and checked by reading, not compiled, not run on hardware
//
#include <hip/hip_runtime.h>
#include <hip/hip_bf16.h>

// ============================================================================
// ViT-S/16 forward for MI455X (gfx1250, wave32, WMMA bf16, fp32 accumulate).
// GEMM: 128x128x32 block tiles, 8 waves, fragment-order LDS staging so that
// every fragment load is 2x ds_load_b128 and every global stage is b128.
// B operands are always stored transposed: Bt[n][k], ldb = padded K.
// ============================================================================

typedef __bf16 bf16;
typedef __bf16 v16bf __attribute__((ext_vector_type(16)));
typedef float  v8f   __attribute__((ext_vector_type(8)));

union Frag16 { v16bf v; bf16 e[16]; };
union FragC  { v8f   v; float e[8]; };

__device__ __forceinline__ bf16 f2bf(float f) {
  unsigned u = __builtin_bit_cast(unsigned, f);
  unsigned r = (u + 0x7FFFu + ((u >> 16) & 1u)) >> 16;   // round-to-nearest-even
  return __builtin_bit_cast(bf16, (unsigned short)r);
}

// ViT constants
#define BATCH   16
#define CIN     3
#define IMGSZ   384
#define PSZ     16
#define GRID_   24
#define NPATCH  576
#define NTOK    577
#define NTOKP   640            // tokens padded to a tile multiple
#define PD      768
#define DIM     384
#define DEPTH   12
#define HEADS   6
#define HD      64
#define MLP     1536
#define NCLS    1000
#define BN      (BATCH * NTOK)       // 9232
#define BNP     9344                 // BN padded to 128
#define BH      (BATCH * HEADS)      // 96

// ---------------------------------------------------------------------------
// GEMM: out = alpha * A @ Bt^T (+bias)(+gelu)(+resid)
//   A  : [batch][>=ceil(M/128)*128 rows][lda = Kp]      (bf16, K padded rows safe)
//   Bt : [batch][>=ceil(N/128)*128 rows][ldb = Kp]      (bf16, transposed)
//   No bounds checks on loads (buffers over-allocated & K zero-padded);
//   stores guarded by (row<M && col<Ncol).
// flags: 1=bias  2=GELU(exact erf)  4=residual add     outMode 1 = head merge
// ---------------------------------------------------------------------------
__global__ __launch_bounds__(256)
void gemm_wmma_bf16(const bf16* __restrict__ A, long sA, int lda,
                    const bf16* __restrict__ Bt, long sB, int ldb,
                    const float* __restrict__ bias,
                    const float* __restrict__ resid,
                    float* __restrict__ outF, bf16* __restrict__ outH,
                    long sC, int ldc, int outMode,
                    int M, int Ncol, int Kp, float alpha, int flags)
{
  // fragment-order staging: slot s (0..255) holds 16 contiguous bf16 = one
  // lane-fragment; s = grp*32 + lane, grp = 16-row (A) / 16-col (B) group.
  __shared__ __align__(32) bf16 Afr[256 * 16];   // 8 KB
  __shared__ __align__(32) bf16 Bfr[256 * 16];   // 8 KB

  const int tid  = threadIdx.x;
  const int lane = tid & 31;
  const int wave = tid >> 5;          // 0..7
  const int l16  = lane & 15;
  const int half = lane >> 4;
  const int grp  = wave;              // staging group 0..7
  const int wr   = wave >> 1;         // wave tile row (0..3)  -> 32 rows
  const int wc   = wave & 1;          // wave tile col (0..1)  -> 64 cols
  const int z    = blockIdx.z;
  const int m0   = blockIdx.y * 128;
  const int n0   = blockIdx.x * 128;

  const bf16* Az = A  + (long)z * sA;
  const bf16* Bz = Bt + (long)z * sB;

  // per-thread fixed staging addresses (k0 added in loop)
  // A slot: row = m0 + grp*16 + l16 ; elems j<8 at k=8*half+j, j>=8 at k=16+8*half+(j-8)
  const bf16* agp = Az + (long)(m0 + grp * 16 + l16) * lda + 8 * half;
  // B slot: n = n0 + grp*16 + l16 ; elems j<8 at k=8*half+j, j>=8 at k=16+8*half+(j-8)
  const bf16* bgp = Bz + (long)(n0 + grp * 16 + l16) * ldb + 8 * half;
  bf16* aslot = &Afr[(grp * 32 + lane) * 16];
  bf16* bslot = &Bfr[(grp * 32 + lane) * 16];

  uint4 a0 = *(const uint4*)(agp);
  uint4 a1 = *(const uint4*)(agp + 16);
  uint4 b0 = *(const uint4*)(bgp);
  uint4 b1 = *(const uint4*)(bgp + 16);

  FragC acc[2][4];
#pragma unroll
  for (int i = 0; i < 2; ++i)
#pragma unroll
    for (int j = 0; j < 4; ++j)
#pragma unroll
      for (int r = 0; r < 8; ++r) acc[i][j].e[r] = 0.0f;

  for (int k0 = 0; k0 < Kp; k0 += 32) {
    // commit staged registers to LDS
    *(uint4*)(aslot)     = a0;
    *(uint4*)(aslot + 8) = a1;
    *(uint4*)(bslot)     = b0;
    *(uint4*)(bslot + 8) = b1;
    __syncthreads();

    // software pipeline: issue next tile's global loads now
    if (k0 + 32 < Kp) {
      a0 = *(const uint4*)(agp + k0 + 32);
      a1 = *(const uint4*)(agp + k0 + 48);
      b0 = *(const uint4*)(bgp + k0 + 32);
      b1 = *(const uint4*)(bgp + k0 + 48);
    }

    // fragment loads: 2x ds_load_b128 each
    Frag16 af[2], bf_[4];
#pragma unroll
    for (int i = 0; i < 2; ++i)
      af[i].v = *(const v16bf*)&Afr[((wr * 2 + i) * 32 + lane) * 16];
#pragma unroll
    for (int j = 0; j < 4; ++j)
      bf_[j].v = *(const v16bf*)&Bfr[((wc * 4 + j) * 32 + lane) * 16];

#pragma unroll
    for (int i = 0; i < 2; ++i)
#pragma unroll
      for (int j = 0; j < 4; ++j)
        acc[i][j].v = __builtin_amdgcn_wmma_f32_16x16x32_bf16(
            false, af[i].v, false, bf_[j].v,
            (short)0, acc[i][j].v, false, false);
    __syncthreads();
  }

  // ---- epilogue ----
  long base = (outMode == 0)
            ? (long)z * sC
            : (long)(z / HEADS) * NTOK * DIM + (long)(z % HEADS) * HD;
#pragma unroll
  for (int i = 0; i < 2; ++i) {
#pragma unroll
    for (int j = 0; j < 4; ++j) {
      int col = n0 + wc * 64 + j * 16 + l16;
#pragma unroll
      for (int r = 0; r < 8; ++r) {
        int row = m0 + wr * 32 + i * 16 + r + 8 * half;
        if (row < M && col < Ncol) {
          float v = acc[i][j].e[r] * alpha;
          if (flags & 1) v += bias[col];
          if (flags & 2) v = 0.5f * v * (1.0f + erff(v * 0.70710678118654752f));
          long idx = base + (long)row * ldc + col;
          if (flags & 4) v += resid[idx];
          if (outF) outF[idx] = v;
          if (outH) outH[idx] = f2bf(v);
        }
      }
    }
  }
}

// ---------------------------------------------------------------------------
// Tiled transpose + fp32->bf16 convert:  src[R,C] row-major -> dst[C, ldDst]
// dst[n*ldDst + k] = src[k*C + n]; k >= R writes 0 (zero K-padding).
// ---------------------------------------------------------------------------
__global__ __launch_bounds__(256)
void transpose_bf16_k(const float* __restrict__ src, bf16* __restrict__ dst,
                      int R, int C, int ldDst)
{
  __shared__ float tile[32][33];
  int bx = blockIdx.x * 32;   // along C
  int by = blockIdx.y * 32;   // along R (padded to ldDst)
  int tx = threadIdx.x, ty = threadIdx.y;   // 32 x 8
#pragma unroll
  for (int i = 0; i < 32; i += 8) {
    int r = by + ty + i, c = bx + tx;
    tile[ty + i][tx] = (r < R && c < C) ? src[(long)r * C + c] : 0.0f;
  }
  __syncthreads();
#pragma unroll
  for (int i = 0; i < 32; i += 8) {
    int n = bx + ty + i;   // dst row
    int k = by + tx;       // dst col
    if (n < C && k < ldDst) dst[(long)n * ldDst + k] = f2bf(tile[tx][ty + i]);
  }
}

// img [B,3,384,384] -> patches bf16 [B*576, 768]  (p1 p2 c ordering)
__global__ void im2col_bf16_k(const float* __restrict__ img, bf16* __restrict__ out, long total) {
  long i = (long)blockIdx.x * blockDim.x + threadIdx.x;
  if (i >= total) return;
  int pd = (int)(i % PD);
  long t = i / PD;
  int patch = (int)(t % NPATCH);
  int b = (int)(t / NPATCH);
  int c  = pd % CIN;
  int p2 = (pd / CIN) % PSZ;
  int p1 = pd / (CIN * PSZ);
  int hw = patch / GRID_, ww = patch % GRID_;
  int hh = hw * PSZ + p1, wc = ww * PSZ + p2;
  out[i] = f2bf(img[(((long)b * CIN + c) * IMGSZ + hh) * IMGSZ + wc]);
}

// x[b,n,d] = (n==0 ? cls[d] : xp[b,n-1,d]) + pos[n,d]
__global__ void assemble_x_k(const float* __restrict__ xp, const float* __restrict__ pos,
                             const float* __restrict__ cls, float* __restrict__ x, long total) {
  long i = (long)blockIdx.x * blockDim.x + threadIdx.x;
  if (i >= total) return;
  int d = (int)(i % DIM);
  long t = i / DIM;
  int n = (int)(t % NTOK);
  int b = (int)(t / NTOK);
  float v = (n == 0) ? cls[d] : xp[((long)b * NPATCH + (n - 1)) * DIM + d];
  x[i] = v + pos[(long)n * DIM + d];
}

// LayerNorm over last dim (384), fp32 in -> bf16 out
__global__ __launch_bounds__(128)
void layernorm_bf16_k(const float* __restrict__ x, const float* __restrict__ g,
                      const float* __restrict__ b, bf16* __restrict__ out) {
  long row = blockIdx.x;
  const float* xr = x + row * DIM;
  int tid = threadIdx.x;
  float vals[3], s = 0.f, s2 = 0.f;
#pragma unroll
  for (int i = 0; i < 3; ++i) {
    float v = xr[tid + i * 128];
    vals[i] = v; s += v; s2 += v * v;
  }
  __shared__ float red0[128], red1[128];
  red0[tid] = s; red1[tid] = s2; __syncthreads();
  for (int o = 64; o > 0; o >>= 1) {
    if (tid < o) { red0[tid] += red0[tid + o]; red1[tid] += red1[tid + o]; }
    __syncthreads();
  }
  float mean = red0[0] * (1.0f / DIM);
  float var  = red1[0] * (1.0f / DIM) - mean * mean;
  float rstd = rsqrtf(var + 1e-5f);
#pragma unroll
  for (int i = 0; i < 3; ++i) {
    int c = tid + i * 128;
    out[row * DIM + c] = f2bf((vals[i] - mean) * rstd * g[c] + b[c]);
  }
}

// qkv fp32 [B,N,3,H,64] -> q,k bf16 [BH, NTOKP, 64] ; v transposed [BH, 64, NTOKP]
__global__ void split_qkv_k(const float* __restrict__ qkv, bf16* __restrict__ q,
                            bf16* __restrict__ k, bf16* __restrict__ vt, long total) {
  long i = (long)blockIdx.x * blockDim.x + threadIdx.x;
  if (i >= total) return;
  int cc = (int)(i % (3 * DIM));
  long t = i / (3 * DIM);
  int n = (int)(t % NTOK);
  int b = (int)(t / NTOK);
  int s = cc / DIM;
  int h = (cc % DIM) / HD;
  int d = cc % HD;
  bf16 val = f2bf(qkv[i]);
  int bh = b * HEADS + h;
  if (s == 0)      q[((long)bh * NTOKP + n) * HD + d] = val;
  else if (s == 1) k[((long)bh * NTOKP + n) * HD + d] = val;
  else             vt[((long)bh * HD + d) * NTOKP + n] = val;
}

// rowwise softmax over NTOK cols (row stride NTOKP); writes zeros in padding
__global__ __launch_bounds__(128)
void softmax_bf16_k(const float* __restrict__ S, bf16* __restrict__ P) {
  long r = blockIdx.x;                 // 0 .. BH*NTOK-1
  long z = r / NTOK;
  int  n = (int)(r % NTOK);
  const float* sr = S + (z * NTOKP + n) * (long)NTOKP;
  bf16* pr = P + (z * NTOKP + n) * (long)NTOKP;
  int tid = threadIdx.x;
  __shared__ float red[128];
  float mx = -3.4e38f;
  for (int c = tid; c < NTOK; c += 128) mx = fmaxf(mx, sr[c]);
  red[tid] = mx; __syncthreads();
  for (int o = 64; o > 0; o >>= 1) { if (tid < o) red[tid] = fmaxf(red[tid], red[tid + o]); __syncthreads(); }
  mx = red[0]; __syncthreads();
  float sum = 0.f;
  for (int c = tid; c < NTOK; c += 128) sum += expf(sr[c] - mx);
  red[tid] = sum; __syncthreads();
  for (int o = 64; o > 0; o >>= 1) { if (tid < o) red[tid] += red[tid + o]; __syncthreads(); }
  float inv = 1.0f / red[0];
  for (int c = tid; c < NTOKP; c += 128)
    pr[c] = (c < NTOK) ? f2bf(expf(sr[c] - mx) * inv) : f2bf(0.0f);
}

// sigmoid of cls token rows -> bf16 [16, 384]
__global__ void sigmoid_cls_k(const float* __restrict__ x, bf16* __restrict__ out, int total) {
  int i = blockIdx.x * blockDim.x + threadIdx.x;
  if (i >= total) return;
  int d = i % DIM, b = i / DIM;
  float v = x[(long)b * NTOK * DIM + d];
  out[(long)b * DIM + d] = f2bf(1.0f / (1.0f + expf(-v)));
}

// ---------------------------------------------------------------------------
// Host orchestration
// ---------------------------------------------------------------------------
static inline void transposeW(hipStream_t s, const float* src, bf16* dst,
                              int R, int C, int ldDst) {
  dim3 g((C + 31) / 32, (ldDst + 31) / 32);
  transpose_bf16_k<<<g, dim3(32, 8), 0, s>>>(src, dst, R, C, ldDst);
}

static inline void gemm(hipStream_t s,
                        const bf16* A, long sA, int lda,
                        const bf16* Bt, long sB, int ldb,
                        const float* bias, const float* resid,
                        float* outF, bf16* outH, long sC, int ldc, int outMode,
                        int M, int Ncol, int Kp, float alpha, int flags, int batch) {
  dim3 g((Ncol + 127) / 128, (M + 127) / 128, batch);
  gemm_wmma_bf16<<<g, 256, 0, s>>>(A, sA, lda, Bt, sB, ldb, bias, resid,
                                   outF, outH, sC, ldc, outMode, M, Ncol, Kp, alpha, flags);
}

extern "C" void kernel_launch(void* const* d_in, const int* in_sizes, int n_in,
                              void* d_out, int out_size, void* d_ws, size_t ws_size,
                              hipStream_t stream)
{
  (void)in_sizes; (void)n_in; (void)out_size; (void)ws_size;
  const float* img     = (const float*)d_in[0];
  const float* w_patch = (const float*)d_in[1];
  const float* b_patch = (const float*)d_in[2];
  const float* pos     = (const float*)d_in[3];
  const float* cls     = (const float*)d_in[4];
  const float* ln1g    = (const float*)d_in[5];
  const float* ln1b    = (const float*)d_in[6];
  const float* wqkv    = (const float*)d_in[7];
  const float* wout    = (const float*)d_in[8];
  const float* bout    = (const float*)d_in[9];
  const float* ln2g    = (const float*)d_in[10];
  const float* ln2b    = (const float*)d_in[11];
  const float* wff1    = (const float*)d_in[12];
  const float* bff1    = (const float*)d_in[13];
  const float* wff2    = (const float*)d_in[14];
  const float* bff2    = (const float*)d_in[15];
  const float* hw1     = (const float*)d_in[16];
  const float* hb1     = (const float*)d_in[17];
  const float* hw2     = (const float*)d_in[18];
  const float* hb2     = (const float*)d_in[19];
  float* out = (float*)d_out;

  // bump allocator over d_ws (~450 MB used; buffers padded so GEMM loads never
  // need bounds checks -- only stores are guarded)
  char* ws = (char*)d_ws;
  size_t off = 0;
  auto alloc = [&](size_t bytes) -> void* {
    off = (off + 255) & ~(size_t)255;
    void* p = ws + off;
    off += bytes;
    return p;
  };

  // transposed bf16 weights: Bt[n][k], ldb = K (all K multiples of 32)
  bf16* wpatch_t = (bf16*)alloc(sizeof(bf16) * (long)DIM * PD);            // [384][768]
  bf16* wqkv_t   = (bf16*)alloc(sizeof(bf16) * (long)DEPTH * 3 * DIM * DIM); // [1152][384] x12
  bf16* wout_t   = (bf16*)alloc(sizeof(bf16) * (long)DEPTH * DIM * DIM);     // [384][384] x12
  bf16* wff1_t   = (bf16*)alloc(sizeof(bf16) * (long)DEPTH * MLP * DIM);     // [1536][384] x12
  bf16* wff2_t   = (bf16*)alloc(sizeof(bf16) * (long)DEPTH * DIM * MLP);     // [384][1536] x12
  bf16* hw1_t    = (bf16*)alloc(sizeof(bf16) * (long)MLP * DIM);             // [1536][384]
  bf16* hw2_t    = (bf16*)alloc(sizeof(bf16) * (long)1024 * MLP);            // [1024 pad][1536]
  // activations (row-padded for tile reads)
  bf16*  patches = (bf16*)alloc(sizeof(bf16) * (long)BATCH * NPATCH * PD);   // 9216 x 768
  float* xp      = (float*)alloc(sizeof(float) * (long)BATCH * NPATCH * DIM);
  float* x       = (float*)alloc(sizeof(float) * (long)BN * DIM);
  bf16*  h       = (bf16*)alloc(sizeof(bf16) * (long)BNP * DIM);             // 9344 x 384
  float* qkv     = (float*)alloc(sizeof(float) * (long)BN * 3 * DIM);
  bf16*  qb      = (bf16*)alloc(sizeof(bf16) * (long)BH * NTOKP * HD);
  bf16*  kb      = (bf16*)alloc(sizeof(bf16) * (long)BH * NTOKP * HD);
  bf16*  vt      = (bf16*)alloc(sizeof(bf16) * (long)(BH + 1) * HD * NTOKP); // +1 batch slack
  float* S       = (float*)alloc(sizeof(float) * (long)BH * NTOKP * NTOKP);
  bf16*  attnP   = (bf16*)alloc(sizeof(bf16) * (long)BH * NTOKP * NTOKP);
  bf16*  hff     = (bf16*)alloc(sizeof(bf16) * (long)BNP * MLP);             // 9344 x 1536
  bf16*  clsb    = (bf16*)alloc(sizeof(bf16) * (long)128 * DIM);
  bf16*  hh      = (bf16*)alloc(sizeof(bf16) * (long)128 * MLP);

  // ---- weight transpose+convert (fp32 [K,N] -> bf16 [N,K]) ----
  transposeW(stream, w_patch, wpatch_t, PD, DIM, PD);
  for (int L = 0; L < DEPTH; ++L) {
    transposeW(stream, wqkv + (long)L * DIM * 3 * DIM, wqkv_t + (long)L * 3 * DIM * DIM, DIM, 3 * DIM, DIM);
    transposeW(stream, wout + (long)L * DIM * DIM,     wout_t + (long)L * DIM * DIM,     DIM, DIM,     DIM);
    transposeW(stream, wff1 + (long)L * DIM * MLP,     wff1_t + (long)L * MLP * DIM,     DIM, MLP,     DIM);
    transposeW(stream, wff2 + (long)L * MLP * DIM,     wff2_t + (long)L * DIM * MLP,     MLP, DIM,     MLP);
  }
  transposeW(stream, hw1, hw1_t, DIM, MLP, DIM);
  transposeW(stream, hw2, hw2_t, MLP, NCLS, MLP);

  // ---- patch embedding ----
  {
    long tot = (long)BATCH * NPATCH * PD;
    im2col_bf16_k<<<(unsigned)((tot + 255) / 256), 256, 0, stream>>>(img, patches, tot);
  }
  gemm(stream, patches, 0, PD, wpatch_t, 0, PD, b_patch, nullptr,
       xp, nullptr, 0, DIM, 0, BATCH * NPATCH, DIM, PD, 1.0f, 1, 1);
  {
    long tot = (long)BN * DIM;
    assemble_x_k<<<(unsigned)((tot + 255) / 256), 256, 0, stream>>>(xp, pos, cls, x, tot);
  }

  const float SCALE = 0.051031036307982884f;  // 384^-0.5

  // ---- transformer layers ----
  for (int L = 0; L < DEPTH; ++L) {
    const bf16* wq = wqkv_t + (long)L * 3 * DIM * DIM;
    const bf16* wo = wout_t + (long)L * DIM * DIM;
    const bf16* w1 = wff1_t + (long)L * MLP * DIM;
    const bf16* w2 = wff2_t + (long)L * DIM * MLP;

    layernorm_bf16_k<<<BN, 128, 0, stream>>>(x, ln1g + (long)L * DIM, ln1b + (long)L * DIM, h);
    // qkv = h @ Wqkv (no bias)
    gemm(stream, h, 0, DIM, wq, 0, DIM, nullptr, nullptr,
         qkv, nullptr, 0, 3 * DIM, 0, BN, 3 * DIM, DIM, 1.0f, 0, 1);
    {
      long tot = (long)BN * 3 * DIM;
      split_qkv_k<<<(unsigned)((tot + 255) / 256), 256, 0, stream>>>(qkv, qb, kb, vt, tot);
    }
    // S = (q @ k^T) * SCALE
    gemm(stream, qb, (long)NTOKP * HD, HD, kb, (long)NTOKP * HD, HD,
         nullptr, nullptr, S, nullptr, (long)NTOKP * NTOKP, NTOKP, 0,
         NTOK, NTOK, HD, SCALE, 0, BH);
    softmax_bf16_k<<<BH * NTOK, 128, 0, stream>>>(S, attnP);
    // O = attn @ v  (K padded to 640; attnP padding cols are exact zeros)
    gemm(stream, attnP, (long)NTOKP * NTOKP, NTOKP, vt, (long)HD * NTOKP, NTOKP,
         nullptr, nullptr, nullptr, h, 0, DIM, /*outMode*/1,
         NTOK, HD, NTOKP, 1.0f, 0, BH);
    // x = x + O @ Wo + bo
    gemm(stream, h, 0, DIM, wo, 0, DIM, bout + (long)L * DIM, x,
         x, nullptr, 0, DIM, 0, BN, DIM, DIM, 1.0f, 1 | 4, 1);
    layernorm_bf16_k<<<BN, 128, 0, stream>>>(x, ln2g + (long)L * DIM, ln2b + (long)L * DIM, h);
    // hff = gelu(h @ W1 + b1)
    gemm(stream, h, 0, DIM, w1, 0, DIM, bff1 + (long)L * MLP, nullptr,
         nullptr, hff, 0, MLP, 0, BN, MLP, DIM, 1.0f, 1 | 2, 1);
    // x = x + hff @ W2 + b2
    gemm(stream, hff, 0, MLP, w2, 0, MLP, bff2 + (long)L * DIM, x,
         x, nullptr, 0, DIM, 0, BN, DIM, MLP, 1.0f, 1 | 4, 1);
  }

  // ---- head ----
  sigmoid_cls_k<<<(BATCH * DIM + 255) / 256, 256, 0, stream>>>(x, clsb, BATCH * DIM);
  gemm(stream, clsb, 0, DIM, hw1_t, 0, DIM, hb1, nullptr,
       nullptr, hh, 0, MLP, 0, BATCH, MLP, DIM, 1.0f, 1 | 2, 1);
  gemm(stream, hh, 0, MLP, hw2_t, 0, MLP, hb2, nullptr,
       out, nullptr, 0, NCLS, 0, BATCH, NCLS, MLP, 1.0f, 1, 1);
}
